// SambaYBoundaryFullAttentionLayer_40097814675984
// MI455X (gfx1250) — matrix operand order repaired and checked by reading, not compiled, not run on hardware
//
#include <hip/hip_runtime.h>
#include <hip/hip_bf16.h>
#include <math.h>

typedef __bf16 bf16_t;
typedef __attribute__((ext_vector_type(4)))  bf16_t v4bf;
typedef __attribute__((ext_vector_type(8)))  bf16_t v8bf;
typedef __attribute__((ext_vector_type(16))) bf16_t v16bf;
typedef __attribute__((ext_vector_type(4)))  float  v4f;
typedef __attribute__((ext_vector_type(8)))  float  v8f;

#define H_      2048
#define NH_     16
#define NKV_    4
#define HD_     128
#define KVD_    (NKV_ * HD_)    /* 512 */
#define INTER_  8192
#define L_      2048
#define EPS_    1e-6f
#define SM_SCALE 0.08838834764831845f   /* 1/sqrt(128) */

// ---------------------------------------------------------------------------
// WMMA fragment helpers (CDNA5 16x16x32 bf16, wave32)
// A-matrix 16x32: lane m (=lane&15) holds K = {kh..kh+7} U {16+kh..16+kh+7},
//                 kh = (lane&16)?8:0.
// B-matrix 32x16: lane n (=lane&15) holds K = {kh..kh+7} U {16+kh..16+kh+7},
//                 kh = (lane&16)?8:0.
// C/D 16x16 f32:  vgpr r <-> row r + ((lane&16)?8:0), col = lane&15.
// ---------------------------------------------------------------------------

__device__ __forceinline__ v16bf load_a_frag(const bf16_t* p, int lane) {
  const int ka = (lane & 16) ? 8 : 0;
  v8bf lo = *reinterpret_cast<const v8bf*>(p + ka);
  v8bf hi = *reinterpret_cast<const v8bf*>(p + ka + 16);
  v16bf r;
#pragma unroll
  for (int i = 0; i < 8; ++i) { r[i] = lo[i]; r[i + 8] = hi[i]; }
  return r;
}

__device__ __forceinline__ v16bf load_b_frag(const bf16_t* p, int lane) {
  const int kb = (lane & 16) ? 8 : 0;
  v8bf lo = *reinterpret_cast<const v8bf*>(p + kb);
  v8bf hi = *reinterpret_cast<const v8bf*>(p + kb + 16);
  v16bf r;
#pragma unroll
  for (int i = 0; i < 8; ++i) { r[i] = lo[i]; r[i + 8] = hi[i]; }
  return r;
}

__device__ __forceinline__ v8f wmma_bf16(v16bf a, v16bf b, v8f c) {
  return __builtin_amdgcn_wmma_f32_16x16x32_bf16(
      /*neg_a=*/false, a, /*neg_b=*/false, b,
      /*c_mod=*/(short)0, c, /*reuse_a=*/false, /*reuse_b=*/false);
}

// ---------------------------------------------------------------------------
// fp32 -> bf16 conversion, 4 elements/thread (all sizes are multiples of 4)
// ---------------------------------------------------------------------------
__global__ void __launch_bounds__(256)
f32_to_bf16_kernel(const float* __restrict__ in, bf16_t* __restrict__ out, int n4) {
  int i = blockIdx.x * 256 + threadIdx.x;
  if (i < n4) {
    v4f v = reinterpret_cast<const v4f*>(in)[i];
    v4bf b;
#pragma unroll
    for (int j = 0; j < 4; ++j) b[j] = (bf16_t)v[j];
    reinterpret_cast<v4bf*>(out)[i] = b;
  }
}

// ---------------------------------------------------------------------------
// RMSNorm: fp32 [L,H] -> bf16 [L,H]
// ---------------------------------------------------------------------------
__global__ void __launch_bounds__(256)
rmsnorm_bf16_kernel(const float* __restrict__ x, const float* __restrict__ w,
                    bf16_t* __restrict__ out) {
  __shared__ float red[256];
  const int row = blockIdx.x;
  const float* xr = x + (size_t)row * H_;
  float s = 0.f;
  for (int i = threadIdx.x; i < H_; i += 256) { float v = xr[i]; s += v * v; }
  red[threadIdx.x] = s;
  __syncthreads();
  for (int ofs = 128; ofs > 0; ofs >>= 1) {
    if (threadIdx.x < ofs) red[threadIdx.x] += red[threadIdx.x + ofs];
    __syncthreads();
  }
  const float rstd = rsqrtf(red[0] * (1.0f / H_) + EPS_);
  bf16_t* orow = out + (size_t)row * H_;
  for (int i = threadIdx.x; i < H_; i += 256)
    orow[i] = (bf16_t)(xr[i] * rstd * w[i]);
}

// ---------------------------------------------------------------------------
// GEMM: C[M,N] = A[M,K]_bf16 * B[N,K]_bf16^T  (A,B row-major, K mult of 64)
// WG = 256 threads = 8 waves. Each wave: 32 rows x 64 cols (eight 16x16 accs,
// two A-row blocks sharing the same 4 B fragments -> 1.5 loads/WMMA).
// Block tile: 256 x 64.  M must be a multiple of 256 (M = 2048 here).
// Ping-pong double buffering (2x unroll): no register rotation movs; each
// fragment set gets a full phase (8 WMMAs) of latency cover.
// Optional epilogues:
//   residual != null : add residual[M,N] fp32
//   outf != null     : write fp32
//   outb != null     : write bf16 of (val * scale)
// ---------------------------------------------------------------------------
__global__ void __launch_bounds__(256)
gemm_bf16_kernel(const bf16_t* __restrict__ A, const bf16_t* __restrict__ B,
                 int K, int N,
                 const float* __restrict__ residual,
                 float* __restrict__ outf,
                 bf16_t* __restrict__ outb,
                 float scale) {
  const int lane = threadIdx.x & 31;
  const int wv   = threadIdx.x >> 5;
  const int l15  = lane & 15;
  const int hi8  = (lane & 16) ? 8 : 0;
  const int row0 = blockIdx.y * 256 + wv * 32;   // two 16-row blocks
  const int col0 = blockIdx.x * 64;

  const bf16_t* a0 = A + (size_t)(row0 + l15) * K;
  const bf16_t* a1 = A + (size_t)(row0 + 16 + l15) * K;
  const bf16_t* b_ptr[4];
#pragma unroll
  for (int j = 0; j < 4; ++j)
    b_ptr[j] = B + (size_t)(col0 + j * 16 + l15) * K;

  v8f acc0[4] = {};
  v8f acc1[4] = {};

  // prologue: set0 <- k=0, set1 <- k=32
  v16bf aA0 = load_a_frag(a0, lane);
  v16bf aB0 = load_a_frag(a1, lane);
  v16bf b0[4];
#pragma unroll
  for (int j = 0; j < 4; ++j) b0[j] = load_b_frag(b_ptr[j], lane);
  v16bf aA1 = load_a_frag(a0 + 32, lane);
  v16bf aB1 = load_a_frag(a1 + 32, lane);
  v16bf b1[4];
#pragma unroll
  for (int j = 0; j < 4; ++j) b1[j] = load_b_frag(b_ptr[j] + 32, lane);

  for (int k = 64; k < K; k += 64) {
    // phase 0: consume set0 (data k-64), refill set0 <- k
#pragma unroll
    for (int j = 0; j < 4; ++j) acc0[j] = wmma_bf16(aA0, b0[j], acc0[j]);
#pragma unroll
    for (int j = 0; j < 4; ++j) acc1[j] = wmma_bf16(aB0, b0[j], acc1[j]);
    aA0 = load_a_frag(a0 + k, lane);
    aB0 = load_a_frag(a1 + k, lane);
#pragma unroll
    for (int j = 0; j < 4; ++j) b0[j] = load_b_frag(b_ptr[j] + k, lane);

    // phase 1: consume set1 (data k-32), refill set1 <- k+32
#pragma unroll
    for (int j = 0; j < 4; ++j) acc0[j] = wmma_bf16(aA1, b1[j], acc0[j]);
#pragma unroll
    for (int j = 0; j < 4; ++j) acc1[j] = wmma_bf16(aB1, b1[j], acc1[j]);
    aA1 = load_a_frag(a0 + k + 32, lane);
    aB1 = load_a_frag(a1 + k + 32, lane);
#pragma unroll
    for (int j = 0; j < 4; ++j) b1[j] = load_b_frag(b_ptr[j] + k + 32, lane);
  }

  // epilogue: drain both sets
#pragma unroll
  for (int j = 0; j < 4; ++j) acc0[j] = wmma_bf16(aA0, b0[j], acc0[j]);
#pragma unroll
  for (int j = 0; j < 4; ++j) acc1[j] = wmma_bf16(aB0, b0[j], acc1[j]);
#pragma unroll
  for (int j = 0; j < 4; ++j) acc0[j] = wmma_bf16(aA1, b1[j], acc0[j]);
#pragma unroll
  for (int j = 0; j < 4; ++j) acc1[j] = wmma_bf16(aB1, b1[j], acc1[j]);

#pragma unroll
  for (int j = 0; j < 4; ++j) {
    const int col = col0 + j * 16 + l15;
#pragma unroll
    for (int r = 0; r < 8; ++r) {
      {
        const int row = row0 + hi8 + r;
        float v = acc0[j][r];
        if (residual) v += residual[(size_t)row * N + col];
        if (outf) outf[(size_t)row * N + col] = v;
        if (outb) outb[(size_t)row * N + col] = (bf16_t)(v * scale);
      }
      {
        const int row = row0 + 16 + hi8 + r;
        float v = acc1[j][r];
        if (residual) v += residual[(size_t)row * N + col];
        if (outf) outf[(size_t)row * N + col] = v;
        if (outb) outb[(size_t)row * N + col] = (bf16_t)(v * scale);
      }
    }
  }
}

// ---------------------------------------------------------------------------
// Fused gate/up GEMM + SiLU(gate)*up epilogue -> bf16 act [L, INTER]
// Ping-pong double buffering (2x unroll), 16 rows x 64 cols, two B sets.
// ---------------------------------------------------------------------------
__global__ void __launch_bounds__(256)
gemm_gateup_kernel(const bf16_t* __restrict__ A, const bf16_t* __restrict__ Bg,
                   const bf16_t* __restrict__ Bu, bf16_t* __restrict__ out) {
  const int K = H_, N = INTER_;
  const int lane = threadIdx.x & 31;
  const int wv   = threadIdx.x >> 5;
  const int l15  = lane & 15;
  const int hi8  = (lane & 16) ? 8 : 0;
  const int row0 = blockIdx.y * 128 + wv * 16;
  const int col0 = blockIdx.x * 64;

  const bf16_t* a_ptr = A + (size_t)(row0 + l15) * K;
  const bf16_t* bg_ptr[4];
  const bf16_t* bu_ptr[4];
#pragma unroll
  for (int j = 0; j < 4; ++j) {
    bg_ptr[j] = Bg + (size_t)(col0 + j * 16 + l15) * K;
    bu_ptr[j] = Bu + (size_t)(col0 + j * 16 + l15) * K;
  }

  v8f accg[4] = {};
  v8f accu[4] = {};

  // prologue: set0 <- k=0, set1 <- k=32
  v16bf a0 = load_a_frag(a_ptr, lane);
  v16bf g0[4], u0[4];
#pragma unroll
  for (int j = 0; j < 4; ++j) {
    g0[j] = load_b_frag(bg_ptr[j], lane);
    u0[j] = load_b_frag(bu_ptr[j], lane);
  }
  v16bf a1 = load_a_frag(a_ptr + 32, lane);
  v16bf g1[4], u1[4];
#pragma unroll
  for (int j = 0; j < 4; ++j) {
    g1[j] = load_b_frag(bg_ptr[j] + 32, lane);
    u1[j] = load_b_frag(bu_ptr[j] + 32, lane);
  }

  for (int k = 64; k < K; k += 64) {
    // phase 0
#pragma unroll
    for (int j = 0; j < 4; ++j) {
      accg[j] = wmma_bf16(a0, g0[j], accg[j]);
      accu[j] = wmma_bf16(a0, u0[j], accu[j]);
    }
    a0 = load_a_frag(a_ptr + k, lane);
#pragma unroll
    for (int j = 0; j < 4; ++j) {
      g0[j] = load_b_frag(bg_ptr[j] + k, lane);
      u0[j] = load_b_frag(bu_ptr[j] + k, lane);
    }
    // phase 1
#pragma unroll
    for (int j = 0; j < 4; ++j) {
      accg[j] = wmma_bf16(a1, g1[j], accg[j]);
      accu[j] = wmma_bf16(a1, u1[j], accu[j]);
    }
    a1 = load_a_frag(a_ptr + k + 32, lane);
#pragma unroll
    for (int j = 0; j < 4; ++j) {
      g1[j] = load_b_frag(bg_ptr[j] + k + 32, lane);
      u1[j] = load_b_frag(bu_ptr[j] + k + 32, lane);
    }
  }

  // epilogue: drain both sets
#pragma unroll
  for (int j = 0; j < 4; ++j) {
    accg[j] = wmma_bf16(a0, g0[j], accg[j]);
    accu[j] = wmma_bf16(a0, u0[j], accu[j]);
  }
#pragma unroll
  for (int j = 0; j < 4; ++j) {
    accg[j] = wmma_bf16(a1, g1[j], accg[j]);
    accu[j] = wmma_bf16(a1, u1[j], accu[j]);
  }

#pragma unroll
  for (int j = 0; j < 4; ++j) {
    const int col = col0 + j * 16 + l15;
#pragma unroll
    for (int r = 0; r < 8; ++r) {
      const int row = row0 + hi8 + r;
      float g = accg[j][r];
      float u = accu[j][r];
      float s = g / (1.0f + __expf(-g));   // SiLU
      out[(size_t)row * N + col] = (bf16_t)(s * u);
    }
  }
}

// ---------------------------------------------------------------------------
// V transpose: v_bf16 [L, KVD] -> vT_bf16 [KVD, L]
// ---------------------------------------------------------------------------
__global__ void __launch_bounds__(256)
transpose_v_kernel(const bf16_t* __restrict__ v, bf16_t* __restrict__ vT) {
  const int idx = blockIdx.x * 256 + threadIdx.x;   // idx = c*L + l
  const int c = idx >> 11;           // L_ = 2048
  const int l = idx & (L_ - 1);
  vT[idx] = v[(size_t)l * KVD_ + c];
}

// ---------------------------------------------------------------------------
// Flash attention (varlen causal, GQA).  One wave per (head, 16-row q-block).
// q_bf16 [L, NH*HD] (pre-scaled by 1/sqrt(HD)), k_bf16 [L, KVD],
// vT_bf16 [KVD, L].  Output: attn_bf16 [L, NH*HD].
// Segment boundaries (cu) and step (32) are all multiples of 32, so each
// 16-row q-block lies in one segment; key loop covers [seg_start, qbase+16).
// ---------------------------------------------------------------------------
__global__ void __launch_bounds__(256)
attn_kernel(const bf16_t* __restrict__ q, const bf16_t* __restrict__ k,
            const bf16_t* __restrict__ vT, const int* __restrict__ cu,
            bf16_t* __restrict__ out) {
  __shared__ bf16_t pbuf[8][16 * 32];   // per-wave 16x32 P tile (1 KB each)

  const int lane  = threadIdx.x & 31;
  const int wv    = threadIdx.x >> 5;
  const int l15   = lane & 15;
  const int hi8   = (lane & 16) ? 8 : 0;
  const int gw    = blockIdx.x * 8 + wv;
  const int h     = gw >> 7;            // 128 q-blocks per head
  const int qb    = gw & 127;
  const int qbase = qb * 16;
  const int kvh   = h >> 2;             // repeat_interleave GQA mapping

  int seg = 0;
  if (cu[1] <= qbase) seg = cu[1];
  if (cu[2] <= qbase) seg = cu[2];

  // Q fragments for the whole head-dim (4 chunks of 32), reused all k-blocks
  v16bf qf[4];
#pragma unroll
  for (int c = 0; c < 4; ++c)
    qf[c] = load_a_frag(q + (size_t)(qbase + l15) * (NH_ * HD_) + h * HD_ + c * 32, lane);

  v8f o[8] = {};
  float rmax[8], rsum[8];
#pragma unroll
  for (int r = 0; r < 8; ++r) { rmax[r] = -3.0e38f; rsum[r] = 0.f; }

  bf16_t* pl = &pbuf[wv][0];

  for (int kb = seg; kb < qbase + 16; kb += 32) {
    // ---- issue all 8 K fragments, then 8 S-WMMAs (max loads in flight) ----
    v16bf kf[8];
#pragma unroll
    for (int c = 0; c < 4; ++c)
      kf[c] = load_b_frag(k + (size_t)(kb + l15) * KVD_ + kvh * HD_ + c * 32, lane);
#pragma unroll
    for (int c = 0; c < 4; ++c)
      kf[4 + c] = load_b_frag(k + (size_t)(kb + 16 + l15) * KVD_ + kvh * HD_ + c * 32, lane);

    v8f s0 = {};
    v8f s1 = {};
#pragma unroll
    for (int c = 0; c < 4; ++c) s0 = wmma_bf16(qf[c], kf[c], s0);
#pragma unroll
    for (int c = 0; c < 4; ++c) s1 = wmma_bf16(qf[c], kf[4 + c], s1);

    // ---- causal mask (only trailing blocks can violate k <= q) ----
    if (kb + 32 > qbase) {
#pragma unroll
      for (int r = 0; r < 8; ++r) {
        const int qrow = qbase + hi8 + r;
        if (kb + l15 > qrow)      s0[r] = -3.0e38f;
        if (kb + 16 + l15 > qrow) s1[r] = -3.0e38f;
      }
    }

    // ---- online softmax: row stats via shfl over the 16-lane half ----
    float tm[8];
#pragma unroll
    for (int r = 0; r < 8; ++r) {
      float t = fmaxf(s0[r], s1[r]);
      t = fmaxf(t, __shfl_xor(t, 1, 32));
      t = fmaxf(t, __shfl_xor(t, 2, 32));
      t = fmaxf(t, __shfl_xor(t, 4, 32));
      t = fmaxf(t, __shfl_xor(t, 8, 32));
      tm[r] = t;
    }
    float al[8];
#pragma unroll
    for (int r = 0; r < 8; ++r) {
      float nm = fmaxf(rmax[r], tm[r]);
      al[r] = __expf(rmax[r] - nm);
      rmax[r] = nm;
    }
    float p0[8], p1[8];
#pragma unroll
    for (int r = 0; r < 8; ++r) {
      p0[r] = __expf(s0[r] - rmax[r]);
      p1[r] = __expf(s1[r] - rmax[r]);
      float t = p0[r] + p1[r];
      t += __shfl_xor(t, 1, 32);
      t += __shfl_xor(t, 2, 32);
      t += __shfl_xor(t, 4, 32);
      t += __shfl_xor(t, 8, 32);
      rsum[r] = rsum[r] * al[r] + t;
    }
#pragma unroll
    for (int d = 0; d < 8; ++d)
#pragma unroll
      for (int r = 0; r < 8; ++r) o[d][r] *= al[r];

    // ---- C-layout -> A-layout for P via per-wave LDS tile ----
#pragma unroll
    for (int r = 0; r < 8; ++r) {
      const int m = hi8 + r;
      pl[m * 32 + l15]      = (bf16_t)p0[r];
      pl[m * 32 + 16 + l15] = (bf16_t)p1[r];
    }

    // issue V fragment loads so they overlap the LDS round-trip
    v16bf vf[8];
#pragma unroll
    for (int d = 0; d < 8; ++d)
      vf[d] = load_b_frag(vT + (size_t)(kvh * HD_ + d * 16 + l15) * L_ + kb, lane);

    v16bf pf = load_a_frag(pl + l15 * 32, lane);   // same-wave LDS: DS in-order

    // ---- O += P * V ----
#pragma unroll
    for (int d = 0; d < 8; ++d) o[d] = wmma_bf16(pf, vf[d], o[d]);
  }

  // ---- normalize and write bf16 ----
#pragma unroll
  for (int d = 0; d < 8; ++d) {
#pragma unroll
    for (int r = 0; r < 8; ++r) {
      const int row = qbase + hi8 + r;
      const int col = h * HD_ + d * 16 + l15;
      out[(size_t)row * (NH_ * HD_) + col] = (bf16_t)(o[d][r] / rsum[r]);
    }
  }
}

// ---------------------------------------------------------------------------
// Host-side orchestration
// ---------------------------------------------------------------------------
extern "C" void kernel_launch(void* const* d_in, const int* in_sizes, int n_in,
                              void* d_out, int out_size, void* d_ws, size_t ws_size,
                              hipStream_t stream) {
  (void)in_sizes; (void)n_in; (void)out_size; (void)ws_size;

  const float* hidden = (const float*)d_in[0];
  const int*   cu     = (const int*)d_in[1];
  const float* ln1    = (const float*)d_in[2];
  const float* ln2    = (const float*)d_in[3];
  const float* qw     = (const float*)d_in[4];
  const float* kw     = (const float*)d_in[5];
  const float* vw     = (const float*)d_in[6];
  const float* ow     = (const float*)d_in[7];
  const float* gw     = (const float*)d_in[8];
  const float* uw     = (const float*)d_in[9];
  const float* dw     = (const float*)d_in[10];

  float* out_hidden = (float*)d_out;                       // [L, H]
  float* out_key    = out_hidden + (size_t)L_ * H_;        // [L, KVD]
  float* out_val    = out_key + (size_t)L_ * KVD_;         // [L, KVD]

  char* ws = (char*)d_ws;
  size_t off = 0;
  auto carve = [&](size_t bytes) -> char* {
    char* p = ws + off;
    off += (bytes + 255) & ~(size_t)255;
    return p;
  };
  bf16_t* wq    = (bf16_t*)carve((size_t)H_ * H_ * 2);
  bf16_t* wk    = (bf16_t*)carve((size_t)KVD_ * H_ * 2);
  bf16_t* wvv   = (bf16_t*)carve((size_t)KVD_ * H_ * 2);
  bf16_t* wo    = (bf16_t*)carve((size_t)H_ * H_ * 2);
  bf16_t* wgt   = (bf16_t*)carve((size_t)INTER_ * H_ * 2);
  bf16_t* wup   = (bf16_t*)carve((size_t)INTER_ * H_ * 2);
  bf16_t* wdn   = (bf16_t*)carve((size_t)H_ * INTER_ * 2);
  bf16_t* normed= (bf16_t*)carve((size_t)L_ * H_ * 2);     // reused for normed2
  bf16_t* qbf   = (bf16_t*)carve((size_t)L_ * H_ * 2);
  bf16_t* kbf   = (bf16_t*)carve((size_t)L_ * KVD_ * 2);
  bf16_t* vbf   = (bf16_t*)carve((size_t)L_ * KVD_ * 2);
  bf16_t* vTb   = (bf16_t*)carve((size_t)KVD_ * L_ * 2);
  bf16_t* attnb = (bf16_t*)carve((size_t)L_ * H_ * 2);
  bf16_t* actb  = (bf16_t*)carve((size_t)L_ * INTER_ * 2);
  float*  hid1  = (float*) carve((size_t)L_ * H_ * 4);

  auto cvt = [&](const float* src, bf16_t* dst, size_t n) {
    const unsigned n4 = (unsigned)(n / 4);
    f32_to_bf16_kernel<<<(n4 + 255) / 256, 256, 0, stream>>>(src, dst, (int)n4);
  };
  cvt(qw, wq,  (size_t)H_ * H_);
  cvt(kw, wk,  (size_t)KVD_ * H_);
  cvt(vw, wvv, (size_t)KVD_ * H_);
  cvt(ow, wo,  (size_t)H_ * H_);
  cvt(gw, wgt, (size_t)INTER_ * H_);
  cvt(uw, wup, (size_t)INTER_ * H_);
  cvt(dw, wdn, (size_t)H_ * INTER_);

  // ---- attention block ----
  rmsnorm_bf16_kernel<<<L_, 256, 0, stream>>>(hidden, ln1, normed);

  // q = normed @ q_w^T, pre-scaled by 1/sqrt(HD), bf16 only
  gemm_bf16_kernel<<<dim3(H_ / 64, L_ / 256), 256, 0, stream>>>(
      normed, wq, H_, H_, nullptr, nullptr, qbf, SM_SCALE);
  // k/v: fp32 -> d_out (key_states/value_states) AND bf16 for attention
  gemm_bf16_kernel<<<dim3(KVD_ / 64, L_ / 256), 256, 0, stream>>>(
      normed, wk, H_, KVD_, nullptr, out_key, kbf, 1.0f);
  gemm_bf16_kernel<<<dim3(KVD_ / 64, L_ / 256), 256, 0, stream>>>(
      normed, wvv, H_, KVD_, nullptr, out_val, vbf, 1.0f);

  transpose_v_kernel<<<(KVD_ * L_) / 256, 256, 0, stream>>>(vbf, vTb);

  attn_kernel<<<(NH_ * (L_ / 16)) / 8, 256, 0, stream>>>(qbf, kbf, vTb, cu, attnb);

  // o-proj + residual -> hid1 (fp32)
  gemm_bf16_kernel<<<dim3(H_ / 64, L_ / 256), 256, 0, stream>>>(
      attnb, wo, H_, H_, hidden, hid1, nullptr, 1.0f);

  // ---- MLP block ----
  rmsnorm_bf16_kernel<<<L_, 256, 0, stream>>>(hid1, ln2, normed);

  gemm_gateup_kernel<<<dim3(INTER_ / 64, L_ / 128), 256, 0, stream>>>(
      normed, wgt, wup, actb);

  // down-proj + residual -> final hidden (fp32, d_out)
  gemm_bf16_kernel<<<dim3(H_ / 64, L_ / 256), 256, 0, stream>>>(
      actb, wdn, INTER_, H_, hid1, out_hidden, nullptr, 1.0f);
}
